// ModernAttentionPlatonic_7928509629010
// MI455X (gfx1250) — matrix operand order repaired and checked by reading, not compiled, not run on hardware
//
#include <hip/hip_runtime.h>
#include <hip/hip_bf16.h>
#include <stdint.h>

typedef __bf16 bf16;
typedef __attribute__((ext_vector_type(16))) __bf16 v16bf;
typedef __attribute__((ext_vector_type(8)))  float  v8f;

#define G_      24
#define NHEAD   96      // G*H
#define B_      2
#define NSEQ    512
#define NTOK    1024
#define DMODEL  3072    // G*H*32
#define DIN     768     // G*32

// ---------------------------------------------------------------------------
// Octahedral rotation group: 24 matrices + left-division table, generated at
// compile time in the exact itertools order of the reference.
// ---------------------------------------------------------------------------
struct Octa { int T[G_][G_]; float R[G_][9]; };

constexpr Octa make_octa() {
  Octa o = {};
  int perms[6][3] = {{0,1,2},{0,2,1},{1,0,2},{1,2,0},{2,0,1},{2,1,0}};
  int sgn[8][3]   = {{1,1,1},{1,1,-1},{1,-1,1},{1,-1,-1},
                     {-1,1,1},{-1,1,-1},{-1,-1,1},{-1,-1,-1}};
  int M[G_][3][3] = {};
  int cnt = 0;
  for (int p = 0; p < 6; ++p)
    for (int s = 0; s < 8; ++s) {
      int m[3][3] = {};
      for (int i = 0; i < 3; ++i) m[i][perms[p][i]] = sgn[s][i];
      int det = m[0][0]*(m[1][1]*m[2][2]-m[1][2]*m[2][1])
              - m[0][1]*(m[1][0]*m[2][2]-m[1][2]*m[2][0])
              + m[0][2]*(m[1][0]*m[2][1]-m[1][1]*m[2][0]);
      if (det == 1) {
        for (int i = 0; i < 3; ++i)
          for (int j = 0; j < 3; ++j) M[cnt][i][j] = m[i][j];
        ++cnt;
      }
    }
  for (int g = 0; g < G_; ++g)
    for (int i = 0; i < 3; ++i)
      for (int j = 0; j < 3; ++j) o.R[g][i*3+j] = (float)M[g][i][j];
  for (int g = 0; g < G_; ++g)
    for (int gp = 0; gp < G_; ++gp) {
      int P[3][3] = {};
      for (int i = 0; i < 3; ++i)
        for (int j = 0; j < 3; ++j) {
          int acc = 0;
          for (int k = 0; k < 3; ++k) acc += M[g][k][i] * M[gp][k][j]; // R[g]^T R[gp]
          P[i][j] = acc;
        }
      int idx = 0;
      for (int t = 0; t < G_; ++t) {
        bool eq = true;
        for (int i = 0; i < 3; ++i)
          for (int j = 0; j < 3; ++j) if (M[t][i][j] != P[i][j]) eq = false;
        if (eq) { idx = t; break; }
      }
      o.T[g][gp] = idx;
    }
  return o;
}
__constant__ Octa c_octa = make_octa();

union ABf { v16bf v; uint32_t u[8]; };

// ---------------------------------------------------------------------------
// Build T-expanded weight, bf16, layout [col = g*co+o][row = gp*ci+i]
// ---------------------------------------------------------------------------
__global__ __launch_bounds__(256)
void k_build_wbig(const float* __restrict__ W, bf16* __restrict__ out,
                  int ci, int co) {
  int total = G_ * co * G_ * ci;
  int idx = blockIdx.x * 256 + threadIdx.x;
  if (idx >= total) return;
  int rowlen = G_ * ci;
  int col = idx / rowlen;
  int row = idx - col * rowlen;
  int g  = col / co, oo = col - g * co;
  int gp = row / ci, ii = row - gp * ci;
  int t = c_octa.T[g][gp];
  out[idx] = (bf16)W[((size_t)t * ci + ii) * co + oo];
}

__global__ __launch_bounds__(256)
void k_f32_to_bf16(const float* __restrict__ in, bf16* __restrict__ out, int n) {
  int i = blockIdx.x * 256 + threadIdx.x;
  if (i < n) out[i] = (bf16)in[i];
}

// ---------------------------------------------------------------------------
// Generic bf16 WMMA GEMM: C[M][N] = A[M][K] * Bt[N][K]^T, f32 accumulate.
// Wave computes 16x64; register ping-pong double buffering so fragment loads
// for step k+32 overlap the 4 WMMAs of step k (K is a multiple of 64).
// mode 0: f32 row-major  1: bf16 row-major  2: bf16 [b][head][c][key] (for V)
// ---------------------------------------------------------------------------
__global__ __launch_bounds__(128)
void k_gemm(const bf16* __restrict__ A, const bf16* __restrict__ Bt,
            float* __restrict__ Cf, bf16* __restrict__ Cb,
            int N, int K, int mode) {
  const int wave = threadIdx.x >> 5;
  const int lane = threadIdx.x & 31;
  const int r    = lane & 15;
  const int hi   = lane >> 4;
  const int m0     = blockIdx.x * 16;
  const int n_base = blockIdx.y * 256 + wave * 64;

  v8f acc[4] = {{}, {}, {}, {}};
  const bf16* arow  = A  + (size_t)(m0 + r) * K + hi * 16;
  const bf16* bcol0 = Bt + (size_t)(n_base + r) * K + hi * 16;

  auto loadA = [&](int k0, ABf& a) {
#pragma unroll
    for (int j = 0; j < 4; ++j) {
      a.u[j]     = *(const uint32_t*)(arow + k0 + 2 * j);
      a.u[4 + j] = *(const uint32_t*)(arow + k0 + 8 + 2 * j);
    }
  };
  auto loadB = [&](int k0, ABf* b) {
#pragma unroll
    for (int nt = 0; nt < 4; ++nt) {
      const bf16* p = bcol0 + (size_t)nt * 16 * K + k0;
#pragma unroll
      for (int j = 0; j < 8; ++j) b[nt].u[j] = *(const uint32_t*)(p + 2 * j);
    }
  };
  auto mma = [&](ABf& a, ABf* b) {
#pragma unroll
    for (int nt = 0; nt < 4; ++nt)
      acc[nt] = __builtin_amdgcn_wmma_f32_16x16x32_bf16(
          false, a.v, false, b[nt].v, (short)0, acc[nt], false, false);
  };

  ABf aA, aB;
  ABf bA[4], bB[4];
  loadA(0, aA);
  loadB(0, bA);
  for (int k0 = 0; k0 < K; k0 += 64) {
    loadA(k0 + 32, aB);
    loadB(k0 + 32, bB);
    mma(aA, bA);                               // overlaps the loads above
    if (k0 + 64 < K) { loadA(k0 + 64, aA); loadB(k0 + 64, bA); }
    mma(aB, bB);
  }

#pragma unroll
  for (int nt = 0; nt < 4; ++nt) {
#pragma unroll
    for (int v = 0; v < 8; ++v) {
      int row = m0 + v + hi * 8;            // C layout: M = v (+8 for hi lanes)
      int col = n_base + nt * 16 + r;       // N = lane&15
      float val = acc[nt][v];
      if (mode == 0) {
        Cf[(size_t)row * N + col] = val;
      } else if (mode == 1) {
        Cb[(size_t)row * N + col] = (bf16)val;
      } else {
        int bb = row >> 9, n = row & 511;   // token -> (b, n)
        int head = col >> 5, c = col & 31;  // col = g*128+h*32+c
        Cb[((size_t)((bb * NHEAD + head) * 32 + c)) * NSEQ + n] = (bf16)val;
      }
    }
  }
}

// ---------------------------------------------------------------------------
// Fused RMSNorm (joint over G x 128) + sequence RoPE + platonic RoPE + bf16.
// Token row is staged Global->LDS once via async-to-LDS (ASYNCcnt), so both
// the reduction pass and the scatter-heavy quad pass read LDS, not L2.
// ---------------------------------------------------------------------------
__global__ __launch_bounds__(256)
void k_rmsrope(const float* __restrict__ xr, const float* __restrict__ coords,
               const int* __restrict__ seq, const float* __restrict__ nw,
               const float* __restrict__ freqs, bf16* __restrict__ outb) {
  const int token = blockIdx.x;
  const int tid = threadIdx.x;
  __shared__ __align__(16) float xs[DMODEL];   // 12 KB token row
  __shared__ float red[256];
  __shared__ float s_inv;

  {
    const float* gsrc = xr + (size_t)token * DMODEL;
#pragma unroll
    for (int i = 0; i < 3; ++i) {              // 256 thr * 3 * 16B = 12 KB
      int e = tid * 4 + i * 1024;
      uint32_t laddr = (uint32_t)(uintptr_t)&xs[e];
      uint64_t gaddr = (uint64_t)(uintptr_t)(gsrc + e);
      asm volatile("global_load_async_to_lds_b128 %0, %1, off"
                   :: "v"(laddr), "v"(gaddr) : "memory");
    }
    asm volatile("s_wait_asynccnt 0x0" ::: "memory");
  }
  __syncthreads();

  float ss = 0.f;
  for (int e = tid; e < DMODEL; e += 256) { float v = xs[e]; ss += v * v; }
  red[tid] = ss;
  __syncthreads();
  for (int s = 128; s > 0; s >>= 1) {
    if (tid < s) red[tid] += red[tid + s];
    __syncthreads();
  }
  if (tid == 0) s_inv = rsqrtf(red[0] * (1.0f / DMODEL) + 1e-6f);
  __syncthreads();
  const float inv = s_inv;
  const float pos = (float)seq[token];
  const float cx = coords[token * 3 + 0];
  const float cy = coords[token * 3 + 1];
  const float cz = coords[token * 3 + 2];
  const float LN16 = 9.210340371976184f / 16.0f;  // ln(10000)/16

  for (int qi = tid; qi < 768; qi += 256) {       // 24*4*8 quads per token
    int g = qi >> 5;
    int rem = qi & 31;
    int h = rem >> 3;
    int t = rem & 7;
    int base = g * 128 + h * 32;
    float x0 = xs[base + 2*t     ] * inv * nw[h*32 + 2*t     ];
    float x1 = xs[base + 2*t + 1 ] * inv * nw[h*32 + 2*t + 1 ];
    float x2 = xs[base + 2*t + 16] * inv * nw[h*32 + 2*t + 16];
    float x3 = xs[base + 2*t + 17] * inv * nw[h*32 + 2*t + 17];

    // sequence RoPE (interleaved pairs t and t+8)
    float th0 = __expf(-(float)t * LN16) * pos;
    float th1 = __expf(-(float)(t + 8) * LN16) * pos;
    float s0, c0, s1, c1;
    __sincosf(th0, &s0, &c0);
    __sincosf(th1, &s1, &c1);
    float y0 = x0 * c0 - x1 * s0;
    float y1 = x1 * c0 + x0 * s0;
    float y2 = x2 * c1 - x3 * s1;
    float y3 = x3 * c1 + x2 * s1;

    // platonic RoPE: ang = coords . (R[g] @ freqs[h,d]), pairs (d, d+16)
    const float* Rg = c_octa.R[g];
    const float* fA = freqs + (size_t)(h * 16 + 2 * t) * 3;
    const float* fB = fA + 3;
    float a0 = cx * (Rg[0]*fA[0] + Rg[1]*fA[1] + Rg[2]*fA[2])
             + cy * (Rg[3]*fA[0] + Rg[4]*fA[1] + Rg[5]*fA[2])
             + cz * (Rg[6]*fA[0] + Rg[7]*fA[1] + Rg[8]*fA[2]);
    float a1 = cx * (Rg[0]*fB[0] + Rg[1]*fB[1] + Rg[2]*fB[2])
             + cy * (Rg[3]*fB[0] + Rg[4]*fB[1] + Rg[5]*fB[2])
             + cz * (Rg[6]*fB[0] + Rg[7]*fB[1] + Rg[8]*fB[2]);
    float sa, ca, sb, cb;
    __sincosf(a0, &sa, &ca);
    __sincosf(a1, &sb, &cb);
    float z0 = y0 * ca - y2 * sa;
    float z2 = y0 * sa + y2 * ca;
    float z1 = y1 * cb - y3 * sb;
    float z3 = y1 * sb + y3 * cb;

    bf16* o = outb + (size_t)token * DMODEL + base;
    o[2*t     ] = (bf16)z0;
    o[2*t + 1 ] = (bf16)z1;
    o[2*t + 16] = (bf16)z2;
    o[2*t + 17] = (bf16)z3;
  }
}

// ---------------------------------------------------------------------------
// Flash attention: one wave per (b, head, 16-query block); 32 keys per step,
// ping-pong double buffering of K/V fragments (NSEQ is a multiple of 64).
// q/k: bf16 [token][head*32+c]; vT: bf16 [b][head][c][key]; out ob bf16.
// ---------------------------------------------------------------------------
__global__ __launch_bounds__(128)
void k_attn(const bf16* __restrict__ qb, const bf16* __restrict__ kb,
            const bf16* __restrict__ vT, bf16* __restrict__ ob) {
  __shared__ bf16 Plds[4][16][40];   // per-wave P transpose buffer (padded)
  const int wave = threadIdx.x >> 5;
  const int lane = threadIdx.x & 31;
  const int r  = lane & 15;
  const int hi = lane >> 4;
  const int wid    = blockIdx.x * 4 + wave;
  const int head_t = wid >> 5;            // 32 query blocks per head
  const int qblk   = wid & 31;
  const int b    = head_t / NHEAD;
  const int head = head_t % NHEAD;
  const int q0   = qblk * 16;

  const bf16* Qbase = qb + (size_t)(b * NSEQ) * DMODEL + head * 32;
  const bf16* Kbase = kb + (size_t)(b * NSEQ) * DMODEL + head * 32;
  const bf16* Vbase = vT + (size_t)((b * NHEAD + head) * 32) * NSEQ;

  ABf qa;  // Q 16x32 A-fragment, reused for the whole KV loop
  {
    const bf16* qr = Qbase + (size_t)(q0 + r) * DMODEL + hi * 16;
#pragma unroll
    for (int j = 0; j < 4; ++j) {
      qa.u[j]     = *(const uint32_t*)(qr + 2 * j);
      qa.u[4 + j] = *(const uint32_t*)(qr + 8 + 2 * j);
    }
  }

  float m[8], l[8];
#pragma unroll
  for (int v = 0; v < 8; ++v) { m[v] = -3.0e38f; l[v] = 0.f; }
  v8f o0 = {}, o1 = {};
  const float scale = 0.17677669529663687f;  // 1/sqrt(32)

  auto loadK = [&](int kc, ABf& b0, ABf& b1) {
    const bf16* kr0 = Kbase + (size_t)(kc + r) * DMODEL + hi * 16;
    const bf16* kr1 = kr0 + (size_t)16 * DMODEL;
#pragma unroll
    for (int j = 0; j < 8; ++j) {
      b0.u[j] = *(const uint32_t*)(kr0 + 2 * j);
      b1.u[j] = *(const uint32_t*)(kr1 + 2 * j);
    }
  };
  auto loadV = [&](int kc, ABf& v0, ABf& v1) {
    const bf16* vr0 = Vbase + (size_t)r * NSEQ + kc + hi * 16;
    const bf16* vr1 = vr0 + (size_t)16 * NSEQ;
#pragma unroll
    for (int j = 0; j < 8; ++j) {
      v0.u[j] = *(const uint32_t*)(vr0 + 2 * j);
      v1.u[j] = *(const uint32_t*)(vr1 + 2 * j);
    }
  };

  auto step = [&](ABf& b0, ABf& b1, ABf& vb0, ABf& vb1) {
    v8f s0v = {}, s1v = {};
    s0v = __builtin_amdgcn_wmma_f32_16x16x32_bf16(false, qa.v, false, b0.v,
                                                  (short)0, s0v, false, false);
    s1v = __builtin_amdgcn_wmma_f32_16x16x32_bf16(false, qa.v, false, b1.v,
                                                  (short)0, s1v, false, false);
    // online softmax; row M=v+8*hi lives across the 16 lanes of a half-group
#pragma unroll
    for (int v = 0; v < 8; ++v) {
      float sA = s0v[v] * scale, sB = s1v[v] * scale;
      float mx = fmaxf(sA, sB);
#pragma unroll
      for (int msk = 1; msk <= 8; msk <<= 1)
        mx = fmaxf(mx, __shfl_xor(mx, msk, 32));
      float mn = fmaxf(m[v], mx);
      float alpha = __expf(m[v] - mn);
      float p0 = __expf(sA - mn);
      float p1 = __expf(sB - mn);
      float sum = p0 + p1;
#pragma unroll
      for (int msk = 1; msk <= 8; msk <<= 1) sum += __shfl_xor(sum, msk, 32);
      l[v] = l[v] * alpha + sum;
      m[v] = mn;
      o0[v] *= alpha;
      o1[v] *= alpha;
      Plds[wave][v + hi * 8][r]      = (bf16)p0;  // transpose C->A via LDS
      Plds[wave][v + hi * 8][16 + r] = (bf16)p1;
    }
    asm volatile("s_wait_dscnt 0x0" ::: "memory");

    ABf pa;  // P 16x32 A-fragment
#pragma unroll
    for (int j = 0; j < 4; ++j) {
      pa.u[j]     = *(const uint32_t*)(&Plds[wave][r][hi * 16 + 2 * j]);
      pa.u[4 + j] = *(const uint32_t*)(&Plds[wave][r][hi * 16 + 8 + 2 * j]);
    }
    o0 = __builtin_amdgcn_wmma_f32_16x16x32_bf16(false, pa.v, false, vb0.v,
                                                 (short)0, o0, false, false);
    o1 = __builtin_amdgcn_wmma_f32_16x16x32_bf16(false, pa.v, false, vb1.v,
                                                 (short)0, o1, false, false);
  };

  ABf kA0, kA1, vA0, vA1, kB0, kB1, vB0, vB1;
  loadK(0, kA0, kA1);
  loadV(0, vA0, vA1);
  for (int kc = 0; kc < NSEQ; kc += 64) {
    loadK(kc + 32, kB0, kB1);
    loadV(kc + 32, vB0, vB1);
    step(kA0, kA1, vA0, vA1);                   // overlaps the loads above
    if (kc + 64 < NSEQ) { loadK(kc + 64, kA0, kA1); loadV(kc + 64, vA0, vA1); }
    step(kB0, kB1, vB0, vB1);
  }

#pragma unroll
  for (int v = 0; v < 8; ++v) {
    float invl = 1.0f / l[v];
    int q = q0 + v + hi * 8;
    bf16* orow = ob + (size_t)(b * NSEQ + q) * DMODEL + head * 32;
    orow[r]      = (bf16)(o0[v] * invl);
    orow[16 + r] = (bf16)(o1[v] * invl);
  }
}

// ---------------------------------------------------------------------------
extern "C" void kernel_launch(void* const* d_in, const int* in_sizes, int n_in,
                              void* d_out, int out_size, void* d_ws, size_t ws_size,
                              hipStream_t stream) {
  (void)in_sizes; (void)n_in; (void)out_size; (void)ws_size;
  const float* feat   = (const float*)d_in[0];
  const float* coords = (const float*)d_in[1];
  const int*   seq    = (const int*)  d_in[2];
  const float* Wq     = (const float*)d_in[3];
  const float* Wk     = (const float*)d_in[4];
  const float* Wv     = (const float*)d_in[5];
  const float* Wo     = (const float*)d_in[6];
  const float* qnw    = (const float*)d_in[7];
  const float* knw    = (const float*)d_in[8];
  const float* freqs  = (const float*)d_in[9];

  uint8_t* ws = (uint8_t*)d_ws;
  size_t off = 0;
  auto alloc = [&](size_t bytes) -> void* {
    void* p = ws + off;
    off = (off + bytes + 255) & ~(size_t)255;
    return p;
  };
  const size_t WBIG = (size_t)DMODEL * DIN;         // 768*3072 elems
  bf16* wbq  = (bf16*)alloc(WBIG * 2);
  bf16* wbk  = (bf16*)alloc(WBIG * 2);
  bf16* wbv  = (bf16*)alloc(WBIG * 2);
  bf16* wbo  = (bf16*)alloc(WBIG * 2);
  bf16* xbf  = (bf16*)alloc((size_t)NTOK * DIN * 2);
  float* qraw = (float*)alloc((size_t)NTOK * DMODEL * 4);
  float* kraw = (float*)alloc((size_t)NTOK * DMODEL * 4);
  bf16* qbf  = (bf16*)alloc((size_t)NTOK * DMODEL * 2);
  bf16* kbf  = (bf16*)alloc((size_t)NTOK * DMODEL * 2);
  bf16* vT   = (bf16*)alloc((size_t)NTOK * DMODEL * 2);
  bf16* obf  = (bf16*)alloc((size_t)NTOK * DMODEL * 2);

  const int nwb = (int)((WBIG + 255) / 256);
  k_build_wbig<<<nwb, 256, 0, stream>>>(Wq, wbq, 32, 128);
  k_build_wbig<<<nwb, 256, 0, stream>>>(Wk, wbk, 32, 128);
  k_build_wbig<<<nwb, 256, 0, stream>>>(Wv, wbv, 32, 128);
  k_build_wbig<<<nwb, 256, 0, stream>>>(Wo, wbo, 128, 32);
  k_f32_to_bf16<<<(NTOK * DIN + 255) / 256, 256, 0, stream>>>(feat, xbf, NTOK * DIN);

  // QKV projections: 1024x768 * 768x3072 (bf16 WMMA, f32 accum)
  dim3 gQKV(NTOK / 16, DMODEL / 256);
  k_gemm<<<gQKV, 128, 0, stream>>>(xbf, wbq, qraw, nullptr, DMODEL, DIN, 0);
  k_gemm<<<gQKV, 128, 0, stream>>>(xbf, wbk, kraw, nullptr, DMODEL, DIN, 0);
  k_gemm<<<gQKV, 128, 0, stream>>>(xbf, wbv, nullptr, vT, DMODEL, DIN, 2);

  k_rmsrope<<<NTOK, 256, 0, stream>>>(qraw, coords, seq, qnw, freqs, qbf);
  k_rmsrope<<<NTOK, 256, 0, stream>>>(kraw, coords, seq, knw, freqs, kbf);

  // 192 heads x 32 query blocks = 6144 waves, 4 waves/block
  k_attn<<<(B_ * NHEAD * (NSEQ / 16)) / 4, 128, 0, stream>>>(qbf, kbf, vT, obf);

  // output projection: 1024x3072 * 3072x768 -> f32 d_out
  dim3 gO(NTOK / 16, DIN / 256);
  k_gemm<<<gO, 128, 0, stream>>>(obf, wbo, (float*)d_out, nullptr, DIN, DMODEL, 0);
}
